// ChebConvLayer_74277164417282
// MI455X (gfx1250) — matrix operand order repaired and checked by reading, not compiled, not run on hardware
//
#include <hip/hip_runtime.h>
#include <stdint.h>

// ChebConv K=3 on gfx1250 (MI455X):
//   out = x@(T0-T2) + L @ ( x@T1 + 2*L@(x@T2) ),  L = (2/lambda)*A - I
//
// ~113 GFLOP vs ~0.6 GB of HBM traffic -> compute-bound; use the matrix pipe
// (v_wmma_f32_16x16x32_bf16, f32 accumulate). All WMMA operands are
// pre-swizzled into fragment layouts (contiguous k per lane half) so the hot loops are pure
// {global_load_b128 x10 + v_wmma x4} per 32-wide k-step: no LDS gathers, no
// sub-word packing, no conversions.
//
// Fragment layouts (wave32, 16x16x32 bf16):
//   A-frag: lane l holds row m=(l&15); 16 bf16 = k: {(l>>4)*16 + 0..15}
//   B-frag: lane l holds col n=(l&15); 16 bf16 = k: {(l>>4)*16 + 0..15}
//   C/D   : vgpr r, lane l -> row r + 8*(l>>4), col (l&15)
// Stored as 32B/lane records => 2 global_load_b128 per fragment.

#define BB   32
#define TT   12
#define NN   1024
#define FINF 64
#define OUTF 64

typedef __attribute__((ext_vector_type(16))) __bf16 v16bf;
typedef __attribute__((ext_vector_type(8)))  float  v8f;

union Frag16 { v16bf v; unsigned short s[16]; uint4 q[2]; };
union Pack8  { uint4 q; unsigned short s[8]; };

__device__ __forceinline__ unsigned short f2bf(float f) {   // native v_cvt_bf16_f32
  union { __bf16 h; unsigned short s; } u; u.h = (__bf16)f; return u.s;
}
__device__ __forceinline__ float bf2f(unsigned short h) {
  union { uint32_t u; float f; } a; a.u = ((uint32_t)h) << 16;
  return a.f;
}

// Swizzled B-operand record offset (in ushorts): buffers Rb/Rc/S are
// [bt][kb(=k/32)][j(=n/16)][lane] x 16 bf16 contiguous.
__device__ __forceinline__ size_t srec(int bt, int kb, int j, int l) {
  return ((((size_t)bt * 32 + kb) * 4 + j) * 32 + l) * 16;
}
// Swizzled A-operand (L) record offset: [b][mt16(=m/16)][kb][lane] x 16 bf16.
__device__ __forceinline__ size_t lrec(int b, int mt16, int kb, int l) {
  return ((((size_t)b * 64 + mt16) * 32 + kb) * 32 + l) * 16;
}

// ---------------------------------------------------------------------------
// Kernel 1: scale[b] = 2 / max(max_row(sum_col A[b]), 1)
// ---------------------------------------------------------------------------
__global__ __launch_bounds__(256) void k_lambda(const float* __restrict__ A,
                                                float* __restrict__ scale) {
  int b = blockIdx.x, tid = threadIdx.x;
  int w = tid >> 5, lane = tid & 31;
  const float* Ab = A + (size_t)b * NN * NN;
  float mx = -3.4e38f;
  for (int row = w; row < NN; row += 8) {
    const float* r = Ab + (size_t)row * NN;
    float s = 0.f;
    for (int c = lane; c < NN; c += 32) s += r[c];
    #pragma unroll
    for (int off = 16; off; off >>= 1) s += __shfl_xor(s, off, 32);
    mx = fmaxf(mx, s);
  }
  __shared__ float red[8];
  if (lane == 0) red[w] = mx;
  __syncthreads();
  if (tid == 0) {
    float m = red[0];
    #pragma unroll
    for (int i = 1; i < 8; ++i) m = fmaxf(m, red[i]);
    scale[b] = 2.0f / fmaxf(m, 1.0f);
  }
}

// ---------------------------------------------------------------------------
// Kernel 2: one-time build of L = sc*A - I as bf16 in A-fragment layout.
// One wave per (b, mt16, kb) record-set; grid = 32*64*32/8 blocks.
// ---------------------------------------------------------------------------
__global__ __launch_bounds__(256) void k_prepL(const float* __restrict__ A,
                                               const float* __restrict__ scale,
                                               unsigned short* __restrict__ Ls) {
  int tid = threadIdx.x;
  int gw = blockIdx.x * 8 + (tid >> 5);     // (b*64+mt16)*32+kb
  int lane = tid & 31, half = lane >> 4, lm = lane & 15;
  int b = gw >> 11;
  int rem = gw & 2047;
  int mt16 = rem >> 5, kb = rem & 31;
  float sc = scale[b];
  int m = mt16 * 16 + lm;
  const float* ap = A + ((size_t)b * NN + m) * NN;
  Frag16 pk;
  #pragma unroll
  for (int g = 0; g < 2; ++g) {
    int k0 = kb * 32 + half * 16 + g * 8;
    float4 p0 = *(const float4*)(ap + k0);
    float4 p1 = *(const float4*)(ap + k0 + 4);
    float vv[8] = {p0.x, p0.y, p0.z, p0.w, p1.x, p1.y, p1.z, p1.w};
    #pragma unroll
    for (int i = 0; i < 8; ++i)
      pk.s[g * 8 + i] = f2bf(vv[i] * sc - ((k0 + i) == m ? 1.0f : 0.0f));
  }
  unsigned short* dst = Ls + ((size_t)gw * 32 + lane) * 16;
  *(uint4*)(dst + 0) = pk.q[0];
  *(uint4*)(dst + 8) = pk.q[1];
}

// ---------------------------------------------------------------------------
// Kernel 3: feature transforms, one pass, shared A-fragment:
//   Ra = x@(T0-T2) -> f32 row-major into d_out
//   Rb = x@T1, Rc = x@T2 -> bf16, B-fragment-swizzled ws buffers
// Theta is swizzled into B-fragment layout in LDS once per block.
// ---------------------------------------------------------------------------
__global__ __launch_bounds__(256) void k_rhs(const float* __restrict__ x,
                                             const float* __restrict__ Theta,
                                             float* __restrict__ Ra,
                                             unsigned short* __restrict__ Rb,
                                             unsigned short* __restrict__ Rc) {
  __shared__ __align__(16) unsigned short lin[3][64][64];   // row-major bf16
  __shared__ __align__(16) unsigned short sw[3][256 * 16];  // B-frag records
  int tid = threadIdx.x;
  for (int i = tid; i < 64 * 64; i += 256) {
    float t0 = Theta[i], t1 = Theta[4096 + i], t2 = Theta[8192 + i];
    ((unsigned short*)lin[0])[i] = f2bf(t0 - t2);
    ((unsigned short*)lin[1])[i] = f2bf(t1);
    ((unsigned short*)lin[2])[i] = f2bf(t2);
  }
  __syncthreads();
  // one-time LDS->LDS swizzle: record r = kb*128 + j*32 + l
  #pragma unroll
  for (int p = 0; p < 3; ++p) {
    int r = tid;
    int kb = r >> 7, j = (r >> 5) & 3, l = r & 31;
    int hk = l >> 4, n = j * 16 + (l & 15);
    Frag16 t;
    #pragma unroll
    for (int i = 0; i < 16; ++i) t.s[i] = lin[p][kb * 32 + hk * 16 + i][n];
    *(uint4*)&sw[p][r * 16 + 0] = t.q[0];
    *(uint4*)&sw[p][r * 16 + 8] = t.q[1];
  }
  __syncthreads();

  int w = tid >> 5, lane = tid & 31, half = lane >> 4, lm = lane & 15;
  size_t rowBase = (size_t)blockIdx.x * 128 + (size_t)w * 16;
  size_t m = rowBase + lm;
  v8f ca[4] = {}, cb[4] = {}, cc[4] = {};

  #pragma unroll
  for (int kb = 0; kb < 2; ++kb) {
    Frag16 af;
    const float* xp = x + m * FINF;
    #pragma unroll
    for (int g = 0; g < 2; ++g) {
      int k0 = kb * 32 + half * 16 + g * 8;
      float4 p0 = *(const float4*)(xp + k0);
      float4 p1 = *(const float4*)(xp + k0 + 4);
      float vv[8] = {p0.x, p0.y, p0.z, p0.w, p1.x, p1.y, p1.z, p1.w};
      #pragma unroll
      for (int i = 0; i < 8; ++i) af.s[g * 8 + i] = f2bf(vv[i]);
    }
    #pragma unroll
    for (int j = 0; j < 4; ++j) {
      int rb16 = (kb * 128 + j * 32 + lane) * 16;
      Frag16 bfr;
      bfr.q[0] = *(const uint4*)&sw[0][rb16 + 0];
      bfr.q[1] = *(const uint4*)&sw[0][rb16 + 8];
      ca[j] = __builtin_amdgcn_wmma_f32_16x16x32_bf16(false, af.v, false, bfr.v,
                                                      (short)0, ca[j], false, false);
      bfr.q[0] = *(const uint4*)&sw[1][rb16 + 0];
      bfr.q[1] = *(const uint4*)&sw[1][rb16 + 8];
      cb[j] = __builtin_amdgcn_wmma_f32_16x16x32_bf16(false, af.v, false, bfr.v,
                                                      (short)0, cb[j], false, false);
      bfr.q[0] = *(const uint4*)&sw[2][rb16 + 0];
      bfr.q[1] = *(const uint4*)&sw[2][rb16 + 8];
      cc[j] = __builtin_amdgcn_wmma_f32_16x16x32_bf16(false, af.v, false, bfr.v,
                                                      (short)0, cc[j], false, false);
    }
  }

  // Epilogue. Ra: plain f32 row-major.  Rb/Rc: direct swizzled b128 stores --
  // this wave's C rows map to contiguous record elements i = 8*half + r.
  int bt = (int)(rowBase >> 10);
  int kb_o = (int)((rowBase & 1023) >> 5);
  int hk = w & 1;
  int l_b = lm + (hk << 4);
  #pragma unroll
  for (int j = 0; j < 4; ++j) {
    int n = j * 16 + lm;
    Pack8 pb, pc;
    #pragma unroll
    for (int r = 0; r < 8; ++r) {
      size_t row = rowBase + r + 8 * half;
      Ra[row * OUTF + n] = ca[j][r];
      pb.s[r] = f2bf(cb[j][r]);
      pc.s[r] = f2bf(cc[j][r]);
    }
    size_t off = srec(bt, kb_o, j, l_b) + (size_t)half * 8;
    *(uint4*)(Rb + off) = pb.q;
    *(uint4*)(Rc + off) = pc.q;
  }
}

// ---------------------------------------------------------------------------
// Kernel 4: S = Rb + 2 * (L @ Rc)   (all operands pre-swizzled; no LDS)
// Grid: B*T*8 blocks; block = 128 rows x 64 cols; k-loop over N=1024.
// ---------------------------------------------------------------------------
__global__ __launch_bounds__(256) void k_agg1(const unsigned short* __restrict__ Ls,
                                              const unsigned short* __restrict__ Rc,
                                              const unsigned short* __restrict__ Rb,
                                              unsigned short* __restrict__ S) {
  int tid = threadIdx.x;
  int bt = blockIdx.x >> 3, mt = blockIdx.x & 7;
  int b = bt / TT;
  int w = tid >> 5, lane = tid & 31, half = lane >> 4, lm = lane & 15;
  int mt16 = mt * 8 + w;
  v8f acc[4] = {};

  for (int kb = 0; kb < 32; ++kb) {
    Frag16 af;
    const unsigned short* ap = Ls + lrec(b, mt16, kb, lane);
    af.q[0] = *(const uint4*)(ap + 0);
    af.q[1] = *(const uint4*)(ap + 8);
    #pragma unroll
    for (int j = 0; j < 4; ++j) {
      Frag16 bfr;
      const unsigned short* bp = Rc + srec(bt, kb, j, lane);
      bfr.q[0] = *(const uint4*)(bp + 0);
      bfr.q[1] = *(const uint4*)(bp + 8);
      acc[j] = __builtin_amdgcn_wmma_f32_16x16x32_bf16(false, af.v, false, bfr.v,
                                                       (short)0, acc[j], false, false);
    }
  }
  // epilogue in swizzled space: 16B vector load Rb, add, 16B vector store S
  int kb_o = mt * 4 + (w >> 1);
  int l_b = lm + ((w & 1) << 4);
  #pragma unroll
  for (int j = 0; j < 4; ++j) {
    size_t off = srec(bt, kb_o, j, l_b) + (size_t)half * 8;
    Pack8 rb; rb.q = *(const uint4*)(Rb + off);
    Pack8 so;
    #pragma unroll
    for (int r = 0; r < 8; ++r) so.s[r] = f2bf(bf2f(rb.s[r]) + 2.0f * acc[j][r]);
    *(uint4*)(S + off) = so.q;
  }
}

// ---------------------------------------------------------------------------
// Kernel 5: out = relu(Ra + L @ S)   (Ra already resident in d_out, f32)
// ---------------------------------------------------------------------------
__global__ __launch_bounds__(256) void k_agg2(const unsigned short* __restrict__ Ls,
                                              const unsigned short* __restrict__ S,
                                              float* __restrict__ out) {
  int tid = threadIdx.x;
  int bt = blockIdx.x >> 3, mt = blockIdx.x & 7;
  int b = bt / TT;
  int w = tid >> 5, lane = tid & 31, half = lane >> 4, lm = lane & 15;
  int mt16 = mt * 8 + w;
  int m0 = mt * 128 + w * 16;
  v8f acc[4] = {};

  for (int kb = 0; kb < 32; ++kb) {
    Frag16 af;
    const unsigned short* ap = Ls + lrec(b, mt16, kb, lane);
    af.q[0] = *(const uint4*)(ap + 0);
    af.q[1] = *(const uint4*)(ap + 8);
    #pragma unroll
    for (int j = 0; j < 4; ++j) {
      Frag16 bfr;
      const unsigned short* bp = S + srec(bt, kb, j, lane);
      bfr.q[0] = *(const uint4*)(bp + 0);
      bfr.q[1] = *(const uint4*)(bp + 8);
      acc[j] = __builtin_amdgcn_wmma_f32_16x16x32_bf16(false, af.v, false, bfr.v,
                                                       (short)0, acc[j], false, false);
    }
  }
  float* Op = out + (size_t)bt * NN * OUTF;
  #pragma unroll
  for (int j = 0; j < 4; ++j) {
    int n = j * 16 + lm;
    #pragma unroll
    for (int r = 0; r < 8; ++r) {
      size_t o = (size_t)(m0 + r + 8 * half) * OUTF + n;
      Op[o] = fmaxf(Op[o] + acc[j][r], 0.0f);
    }
  }
}

// ---------------------------------------------------------------------------
extern "C" void kernel_launch(void* const* d_in, const int* in_sizes, int n_in,
                              void* d_out, int out_size, void* d_ws, size_t ws_size,
                              hipStream_t stream) {
  (void)in_sizes; (void)n_in; (void)out_size; (void)ws_size;
  const float* x     = (const float*)d_in[0];
  const float* A     = (const float*)d_in[1];
  const float* Theta = (const float*)d_in[2];
  float* out = (float*)d_out;

  char* ws = (char*)d_ws;
  const size_t LBYTES = (size_t)BB * NN * NN * sizeof(unsigned short);       // 64 MB
  const size_t RBYTES = (size_t)BB * TT * NN * OUTF * sizeof(unsigned short); // 48 MB
  float*          scl = (float*)ws;                                    // 128 B
  unsigned short* Ls  = (unsigned short*)(ws + 256);
  unsigned short* Rb  = (unsigned short*)(ws + 256 + LBYTES);
  unsigned short* Rc  = (unsigned short*)(ws + 256 + LBYTES + RBYTES);
  unsigned short* S   = (unsigned short*)(ws + 256 + LBYTES + 2 * RBYTES);

  k_lambda<<<BB, 256, 0, stream>>>(A, scl);
  k_prepL <<<(BB * 64 * 32) / 8, 256, 0, stream>>>(A, scl, Ls);
  k_rhs   <<<(BB * TT * NN) / 128, 256, 0, stream>>>(x, Theta, out, Rb, Rc);
  k_agg1  <<<BB * TT * 8, 256, 0, stream>>>(Ls, Rc, Rb, S);
  k_agg2  <<<BB * TT * 8, 256, 0, stream>>>(Ls, S, out);
}
